// AttentionHead_72602127172283
// MI455X (gfx1250) — compile-verified, matrix operand order checked by reading, not run on hardware
//
#include <hip/hip_runtime.h>
#include <hip/hip_bf16.h>

#define BB 8
#define S1D 2048
#define S2D 2048
#define EED 1024
#define PPD 1024
#define QPITCH 1032   // 1024 + 8 bf16 pad -> row stride 2064B = 129*16B (bank-spread)

typedef __attribute__((ext_vector_type(16))) __bf16 v16bf;
typedef __attribute__((ext_vector_type(8)))  __bf16 v8bf;
typedef __attribute__((ext_vector_type(8)))  float  v8f;
typedef __attribute__((ext_vector_type(4)))  int    v4i;

typedef __attribute__((address_space(1))) v4i* as1_v4ip;  // global (prints as __device__*)
typedef __attribute__((address_space(3))) v4i* as3_v4ip;  // LDS

// Build a 16-element bf16 WMMA fragment from two 16-byte chunks.
// Per ISA 7.12.2 (16-bit A/B layout): frag[0..7] = K group (half*8 ..),
// frag[8..15] = K group (16+half*8 ..).
static __device__ inline v16bf frag_from_bf16(const __bf16* p0, const __bf16* p1) {
  v8bf lo = *(const v8bf*)p0;
  v8bf hi = *(const v8bf*)p1;
  return __builtin_shufflevector(lo, hi, 0,1,2,3,4,5,6,7,8,9,10,11,12,13,14,15);
}

static __device__ inline v16bf frag_from_f32(const float* p0, const float* p1) {
  v16bf r;
#pragma unroll
  for (int i = 0; i < 8; ++i) { r[i] = (__bf16)p0[i]; r[i + 8] = (__bf16)p1[i]; }
  return r;
}

// Async global->LDS 16B copy (ASYNCcnt path) with synchronous fallback.
static __device__ inline void copy16_g2l(const __bf16* gsrc, __bf16* ldst) {
#if __has_builtin(__builtin_amdgcn_global_load_async_to_lds_b128)
  __builtin_amdgcn_global_load_async_to_lds_b128(
      (as1_v4ip)gsrc, (as3_v4ip)ldst, 0, 0);
#else
  *(v8bf*)ldst = *(const v8bf*)gsrc;
#endif
}

static __device__ inline void wait_async_copies() {
#if __has_builtin(__builtin_amdgcn_global_load_async_to_lds_b128)
#if __has_builtin(__builtin_amdgcn_s_wait_asynccnt)
  __builtin_amdgcn_s_wait_asynccnt(0);
#else
  asm volatile("s_wait_asynccnt 0" ::: "memory");
#endif
#endif
}

// ---------------------------------------------------------------------------
// Projection: out = X (M x E, f32) @ W^T + b, output bf16.
// mode 0: out row-major (M x P).  mode 1: out transposed per batch (B, P, S2).
// Block = 256 threads = 8 wave32 waves; block tile 32(M) x 256(N);
// wave tile 16 x 64 (four 16x16 f32 accumulators), K stepped by 32.
// ---------------------------------------------------------------------------
__global__ __launch_bounds__(256)
void proj_kernel(const float* __restrict__ X, const float* __restrict__ W,
                 const float* __restrict__ bias, __bf16* __restrict__ out,
                 int mode) {
  const int lane = threadIdx.x & 31;
  const int wv   = threadIdx.x >> 5;
  const int r    = lane & 15;
  const int hh   = lane >> 4;

  const int m0 = blockIdx.x * 32 + (wv & 1) * 16;
  const int n0 = blockIdx.y * 256 + (wv >> 1) * 64;

  const float* arow = X + (size_t)(m0 + r) * EED;
  v8f acc[4] = {};

#pragma unroll 2
  for (int k = 0; k < EED; k += 32) {
    v16bf a = frag_from_f32(arow + k + hh * 8, arow + k + 16 + hh * 8);
#pragma unroll
    for (int nt = 0; nt < 4; ++nt) {
      const float* wrow = W + (size_t)(n0 + nt * 16 + r) * EED; // B[k][n] = W[n][k]
      v16bf bfr = frag_from_f32(wrow + k + hh * 8, wrow + k + 16 + hh * 8);
      acc[nt] = __builtin_amdgcn_wmma_f32_16x16x32_bf16(
          false, a, false, bfr, (short)0, acc[nt], false, false);
    }
  }

  if (mode == 0) {
#pragma unroll
    for (int nt = 0; nt < 4; ++nt) {
      const int n = n0 + nt * 16 + r;
      const float bv = bias[n];
#pragma unroll
      for (int v = 0; v < 8; ++v) {
        const int m = m0 + hh * 8 + v;
        out[(size_t)m * PPD + n] = (__bf16)(acc[nt][v] + bv);
      }
    }
  } else {
#pragma unroll
    for (int nt = 0; nt < 4; ++nt) {
      const int n = n0 + nt * 16 + r;
      const float bv = bias[n];
      const int m = m0 + hh * 8;
      const int batch = m / S2D;
      const int sm = m - batch * S2D;
      v8bf pk;
#pragma unroll
      for (int v = 0; v < 8; ++v) pk[v] = (__bf16)(acc[nt][v] + bv);
      *(v8bf*)(out + ((size_t)batch * PPD + n) * S2D + sm) = pk;
    }
  }
}

// ---------------------------------------------------------------------------
// Flash attention over bf16 projected tensors.
// Grid: (B * S1/32, 2). Block = 256 threads (8 waves).
// Q tile (32x1024 bf16) staged once to dynamic LDS via async global->LDS;
// K/V fragments streamed from L2 with b128 loads. Per 64-row KV tile:
// scores (8 waves x one 16x16 tile, K=1024), online softmax (wave 0),
// then O += P @ V (each wave owns 64 output cols of its 512-col split).
// ---------------------------------------------------------------------------
__global__ __launch_bounds__(256)
void attn_kernel(const __bf16* __restrict__ qp, const __bf16* __restrict__ kp,
                 const __bf16* __restrict__ vpT, float* __restrict__ out) {
  extern __shared__ __align__(16) __bf16 sQ[];   // 32 x QPITCH bf16 (~66 KB)
  __shared__ float  sS[32 * 64];                 // raw scores, f32
  __shared__ __bf16 sP[32 * 64];                 // softmax probs, bf16
  __shared__ float  sCorr[32];                   // per-row rescale factor
  __shared__ float  sStat[32];                   // per-row 1/l at the end

  const int lane = threadIdx.x & 31;
  const int wv   = threadIdx.x >> 5;
  const int r    = lane & 15;
  const int hh   = lane >> 4;

  const int batch  = blockIdx.x >> 6;            // 64 q-tiles per batch
  const int q0     = (blockIdx.x & 63) * 32;
  const int nsplit = blockIdx.y;

  const __bf16* qb = qp  + (size_t)batch * S1D * PPD;
  const __bf16* kb = kp  + (size_t)batch * S2D * PPD;
  const __bf16* vb = vpT + (size_t)batch * PPD * S2D;

  // ---- stage Q tile into LDS (async copies, 16B per call per thread)
  {
    const int row = threadIdx.x >> 3;            // 32 rows, 8 threads each
    const int seg = threadIdx.x & 7;             // 128-element segment
    const __bf16* gs = qb + (size_t)(q0 + row) * PPD + seg * 128;
    __bf16* ls = sQ + row * QPITCH + seg * 128;
#pragma unroll
    for (int c = 0; c < 16; ++c)
      copy16_g2l(gs + c * 8, ls + c * 8);
    wait_async_copies();
  }
  __syncthreads();

  const int sm_t  = (wv & 1) * 16;               // score tile M offset
  const int sn_t  = (wv >> 1) * 16;              // score tile N offset (kv)
  const int ncol0 = nsplit * 512 + wv * 64;      // output column base

  const float scale = 0.03125f;                  // 1/sqrt(1024)

  v8f accO[2][4] = {};
  float m_run = -3.0e38f;
  float l_run = 0.0f;

  const __bf16* qlrow = sQ + (sm_t + r) * QPITCH;  // A rows now in LDS

  for (int kv = 0; kv < S2D; kv += 64) {
    // ---- phase 1: S = Q K^T (each wave one 16x16 tile of the 32x64 block)
    v8f accS = {};
    const __bf16* krow = kb + (size_t)(kv + sn_t + r) * PPD; // B[k][n]=K[n][k]
#pragma unroll 4
    for (int e = 0; e < EED; e += 32) {
      v16bf a  = frag_from_bf16(qlrow + e + hh * 8, qlrow + e + 16 + hh * 8);
      v16bf bk = frag_from_bf16(krow + e + hh * 8, krow + e + 16 + hh * 8);
      accS = __builtin_amdgcn_wmma_f32_16x16x32_bf16(
          false, a, false, bk, (short)0, accS, false, false);
    }
#pragma unroll
    for (int v = 0; v < 8; ++v)
      sS[(sm_t + hh * 8 + v) * 64 + sn_t + r] = accS[v];

    if (kv + 64 < S2D)  // hint next K tile toward the caches
      __builtin_prefetch(kb + (size_t)(kv + 64 + sn_t + r) * PPD, 0, 1);

    __syncthreads();

    // ---- phase 2: online softmax row update (wave 0, one lane per row)
    if (wv == 0) {
      const float* srow = sS + lane * 64;
      float mx = m_run;
#pragma unroll 8
      for (int c = 0; c < 64; ++c) mx = fmaxf(mx, srow[c] * scale);
      const float corr = __expf(m_run - mx);
      float ls = 0.0f;
      __bf16* prow = sP + lane * 64;
#pragma unroll 8
      for (int c = 0; c < 64; ++c) {
        const float p = __expf(srow[c] * scale - mx);
        ls += p;
        prow[c] = (__bf16)p;
      }
      l_run = l_run * corr + ls;
      m_run = mx;
      sCorr[lane] = corr;
    }
    __syncthreads();

    // ---- phase 3: rescale O, then O += P @ V (wave owns 64 output cols)
#pragma unroll
    for (int mt = 0; mt < 2; ++mt)
#pragma unroll
      for (int v = 0; v < 8; ++v) {
        const float c = sCorr[mt * 16 + hh * 8 + v];
#pragma unroll
        for (int nt = 0; nt < 4; ++nt) accO[mt][nt][v] *= c;
      }

#pragma unroll
    for (int ks = 0; ks < 64; ks += 32) {
      v16bf aP[2];
#pragma unroll
      for (int mt = 0; mt < 2; ++mt) {
        const __bf16* pr = sP + (mt * 16 + r) * 64 + ks;
        aP[mt] = frag_from_bf16(pr + hh * 8, pr + 16 + hh * 8);
      }
#pragma unroll
      for (int nt = 0; nt < 4; ++nt) {
        // B[k][n] = V[kv+k][n] = vpT[n][kv+k] -> contiguous along k
        const __bf16* vrow = vb + (size_t)(ncol0 + nt * 16 + r) * S2D + kv + ks;
        v16bf bv = frag_from_bf16(vrow + hh * 8, vrow + 16 + hh * 8);
#pragma unroll
        for (int mt = 0; mt < 2; ++mt)
          accO[mt][nt] = __builtin_amdgcn_wmma_f32_16x16x32_bf16(
              false, aP[mt], false, bv, (short)0, accO[mt][nt], false, false);
      }
    }
    __syncthreads();
  }

  // ---- epilogue: divide by row sums, write f32 output
  if (wv == 0) sStat[lane] = 1.0f / l_run;
  __syncthreads();

  float* ob = out + ((size_t)batch * S1D + q0) * PPD;
#pragma unroll
  for (int mt = 0; mt < 2; ++mt)
#pragma unroll
    for (int v = 0; v < 8; ++v) {
      const int row = mt * 16 + hh * 8 + v;
      const float inv = sStat[row];
#pragma unroll
      for (int nt = 0; nt < 4; ++nt)
        ob[(size_t)row * PPD + ncol0 + nt * 16 + r] = accO[mt][nt][v] * inv;
    }
}

extern "C" void kernel_launch(void* const* d_in, const int* in_sizes, int n_in,
                              void* d_out, int out_size, void* d_ws, size_t ws_size,
                              hipStream_t stream) {
  (void)in_sizes; (void)n_in; (void)out_size; (void)ws_size;
  const float* q = (const float*)d_in[0];
  const float* k = (const float*)d_in[1];
  const float* v = (const float*)d_in[2];
  const float* W = (const float*)d_in[3];
  const float* b = (const float*)d_in[4];
  float* out = (float*)d_out;

  // Workspace: bf16 projected Q (B*S1*P), K (B*S2*P), V^T (B*P*S2) = 96 MB
  __bf16* qp  = (__bf16*)d_ws;
  __bf16* kp  = qp + (size_t)BB * S1D * PPD;
  __bf16* vpT = kp + (size_t)BB * S2D * PPD;

  dim3 pg(BB * S1D / 32, PPD / 256);
  proj_kernel<<<pg, 256, 0, stream>>>(q, W, b, qp, 0);
  proj_kernel<<<pg, 256, 0, stream>>>(k, W, b, kp, 0);
  proj_kernel<<<pg, 256, 0, stream>>>(v, W, b, vpT, 1);

  const size_t qlds = (size_t)32 * QPITCH * sizeof(__bf16);  // ~66 KB dynamic LDS
  dim3 ag(BB * (S1D / 32), 2);
  attn_kernel<<<ag, 256, qlds, stream>>>(qp, kp, vpT, out);
}
